// LRUCell_3393024163915
// MI455X (gfx1250) — matrix operand order repaired and checked by reading, not compiled, not run on hardware
//
#include <hip/hip_runtime.h>
#include <stdint.h>

// Problem constants (match reference)
#define Bv   16
#define Tv   2048
#define Hv   512
#define H4v  2048
#define Ov   512
#define Mv   (Bv * Tv)          // 32768 rows
#define LN_EPS 1e-6f

// GEMM tiling
#define BM   128
#define BN   128
#define BK   64                 // staged K-chunk (2 WMMA k-substeps)
#define APAD 72                 // LDS row stride in bf16 elements (144B, 16B aligned)

typedef __attribute__((ext_vector_type(16))) __bf16 v16bf;
typedef __attribute__((ext_vector_type(8)))  __bf16 v8bf;
typedef __attribute__((ext_vector_type(4)))  __bf16 v4bf;
typedef __attribute__((ext_vector_type(8)))  float  v8f;

// ---------------------------------------------------------------------------
// CDNA5 primitives
// ---------------------------------------------------------------------------
__device__ __forceinline__ v8f wmma_bf16(v16bf a, v16bf b, v8f c) {
  // 8 args: (neg_a, A, neg_b, B, c_mod, C, reuse_a, reuse_b)
  return __builtin_amdgcn_wmma_f32_16x16x32_bf16(false, a, false, b,
                                                 (short)0, c, false, false);
}

// Async global->LDS copy, 16B per lane. Tracked with ASYNCcnt.
__device__ __forceinline__ void async_copy_b128(uint32_t lds_off, const void* g) {
  asm volatile("global_load_async_to_lds_b128 %0, %1, off"
               :: "v"(lds_off), "v"(g) : "memory");
}
__device__ __forceinline__ void wait_async0() {
  asm volatile("s_wait_asynccnt 0x0" ::: "memory");
}

// ---------------------------------------------------------------------------
// Fragment loads.
// A (16x32 bf16, ISA layout): lanes 0-15 row M=lane hold K {0..7,16..23},
//                             lanes 16-31 row M=lane-16 hold K {8..15,24..31}.
// B (32x16 bf16, from [N,K] storage): lanes 0-15 col N=lane hold K {0..7,16..23},
//                             lanes 16-31 hold K {8..15,24..31}.
// ---------------------------------------------------------------------------
__device__ __forceinline__ v16bf load_frag_a_lds(const __bf16* base, int mlocal,
                                                 int kloc) {
  const int lane = threadIdx.x & 31;
  const int r  = lane & 15;
  const int hl = lane >> 4;
  const __bf16* p = base + (mlocal + r) * APAD + kloc + hl * 8;
  union { v16bf v; v8bf h[2]; } f;
  f.h[0] = *(const v8bf*)(p);
  f.h[1] = *(const v8bf*)(p + 16);
  return f.v;
}
__device__ __forceinline__ v16bf load_frag_b_lds(const __bf16* base, int nlocal,
                                                 int kloc) {
  const int lane = threadIdx.x & 31;
  const int n  = lane & 15;
  const int kh = lane >> 4;
  const __bf16* p = base + (nlocal + n) * APAD + kloc + kh * 8;
  union { v16bf v; v8bf h[2]; } f;
  f.h[0] = *(const v8bf*)(p);
  f.h[1] = *(const v8bf*)(p + 16);
  return f.v;
}
// Global variants (used by the fused readout+LN kernel)
__device__ __forceinline__ v16bf load_frag_a(const __bf16* __restrict__ A,
                                             int lda, int m0, int k0) {
  const int lane = threadIdx.x & 31;
  const int r  = lane & 15;
  const int hl = lane >> 4;
  const __bf16* p = A + (size_t)(m0 + r) * lda + (k0 + hl * 8);
  union { v16bf v; v8bf h[2]; } f;
  f.h[0] = *(const v8bf*)(p);
  f.h[1] = *(const v8bf*)(p + 16);
  return f.v;
}
__device__ __forceinline__ v16bf load_frag_b(const __bf16* __restrict__ Wt,
                                             int ldb, int n0, int k0) {
  const int lane = threadIdx.x & 31;
  const int n  = lane & 15;
  const int kh = lane >> 4;
  const __bf16* p = Wt + (size_t)(n0 + n) * ldb + (k0 + kh * 8);
  union { v16bf v; v8bf h[2]; } f;
  f.h[0] = *(const v8bf*)(p);
  f.h[1] = *(const v8bf*)(p + 16);
  return f.v;
}

// ---------------------------------------------------------------------------
// Small prep kernels
// ---------------------------------------------------------------------------
__global__ void prep_gamma_kernel(const float* __restrict__ gamma_log,
                                  float* __restrict__ gamma, int n) {
  int i = blockIdx.x * blockDim.x + threadIdx.x;
  if (i < n) gamma[i] = __expf(gamma_log[i]);
}

__global__ void cvt_f32_bf16_kernel(const float* __restrict__ src,
                                    __bf16* __restrict__ dst, size_t n4) {
  size_t i = (size_t)blockIdx.x * blockDim.x + threadIdx.x;
  size_t stride = (size_t)gridDim.x * blockDim.x;
  for (; i < n4; i += stride) {
    float4 v = ((const float4*)src)[i];
    v4bf o = { (__bf16)v.x, (__bf16)v.y, (__bf16)v.z, (__bf16)v.w };
    ((v4bf*)dst)[i] = o;
  }
}

// W[K,N] f32 -> Wt[N,K] bf16 (32x32 LDS tile transpose), 256 threads.
__global__ void transpose_bf16_kernel(const float* __restrict__ W,
                                      __bf16* __restrict__ Wt, int K, int N) {
  __shared__ float tile[32][33];
  const int kb = blockIdx.x * 32;
  const int nb = blockIdx.y * 32;
  const int tx = threadIdx.x & 31;
  const int ty = threadIdx.x >> 5;        // 0..7
  for (int i = ty; i < 32; i += 8)
    tile[i][tx] = W[(size_t)(kb + i) * N + (nb + tx)];
  __syncthreads();
  for (int i = ty; i < 32; i += 8)
    Wt[(size_t)(nb + i) * K + (kb + tx)] = (__bf16)tile[tx][i];
}

// ---------------------------------------------------------------------------
// LDS-staged, double-buffered WMMA GEMM.
//   C[M,N] = A[M,K](bf16) * Bt[N,K]^T(bf16) + bias
//   EPI 0: f32 out = acc + bias[n]
//   EPI 1: f32 out = scale[n] * (acc + bias[n])     (gamma*(x@B + b))
//   EPI 2: bf16 out = acc + bias[n]
// Block tile 128x128, 8 waves (2Mx4N), wave tile 64x32.
// BK=64 panels staged via GLOBAL_LOAD_ASYNC_TO_LDS_B128, double-buffered:
// 16 WMMAs + ds_loads per staged panel overlap the next panel's async copy.
// ---------------------------------------------------------------------------
template <int EPI>
__global__ __launch_bounds__(256) void wmma_gemm_kernel(
    const __bf16* __restrict__ A, const __bf16* __restrict__ Bt,
    const float* __restrict__ bias, const float* __restrict__ scale,
    float* __restrict__ Cf, __bf16* __restrict__ Cb,
    int K, int lda, int ldb, int ldc) {
  __shared__ __bf16 As[2][BM * APAD];
  __shared__ __bf16 Bs[2][BM * APAD];

  const int tid  = threadIdx.x;
  const int w    = tid >> 5;
  const int lane = tid & 31;
  const int wm   = w & 1;               // 2 waves along M (64 rows each)
  const int wn   = w >> 1;              // 4 waves along N (32 cols each)
  const int m0   = blockIdx.x * BM;
  const int n0   = blockIdx.y * BN;

  // Stage one 128x64 A panel + 128x64 B panel into LDS buffer `buf`.
  // 1024 x 16B chunks per panel; 256 threads -> 4 chunks each per panel.
  auto stage = [&](int buf, int kk) {
#pragma unroll
    for (int c = 0; c < 4; ++c) {
      const int chunk = tid + c * 256;        // 0..1023
      const int row = chunk >> 3;             // 0..127
      const int seg = chunk & 7;              // 16B segment within 128B row
      async_copy_b128((uint32_t)(uintptr_t)&As[buf][row * APAD + seg * 8],
                      A + (size_t)(m0 + row) * lda + kk + seg * 8);
      async_copy_b128((uint32_t)(uintptr_t)&Bs[buf][row * APAD + seg * 8],
                      Bt + (size_t)(n0 + row) * ldb + kk + seg * 8);
    }
  };

  v8f acc[4][2] = {};

  stage(0, 0);
  wait_async0();
  __syncthreads();

  int buf = 0;
  for (int k = 0; k < K; k += BK) {
    if (k + BK < K) stage(buf ^ 1, k + BK);   // overlap next panel copy

#pragma unroll
    for (int ks = 0; ks < BK / 32; ++ks) {
      v16bf af[4], bfr[2];
#pragma unroll
      for (int i = 0; i < 4; ++i)
        af[i] = load_frag_a_lds(&As[buf][0], wm * 64 + i * 16, ks * 32);
#pragma unroll
      for (int j = 0; j < 2; ++j)
        bfr[j] = load_frag_b_lds(&Bs[buf][0], wn * 32 + j * 16, ks * 32);
#pragma unroll
      for (int i = 0; i < 4; ++i)
#pragma unroll
        for (int j = 0; j < 2; ++j)
          acc[i][j] = wmma_bf16(af[i], bfr[j], acc[i][j]);
    }

    wait_async0();        // next panel resident before publishing
    __syncthreads();      // all waves done reading `buf`, next buf visible
    buf ^= 1;
  }

  // C/D layout: VGPR j -> row M = j + 8*(lane>>4); col N = lane&15
  const int cn = lane & 15;
  const int rh = lane >> 4;
#pragma unroll
  for (int tm = 0; tm < 4; ++tm) {
#pragma unroll
    for (int tn = 0; tn < 2; ++tn) {
      const int c  = n0 + wn * 32 + tn * 16 + cn;
      const int r0 = m0 + wm * 64 + tm * 16 + rh * 8;
      const float bv = bias[c];
      const float sv = (EPI == 1) ? scale[c] : 1.0f;
#pragma unroll
      for (int j = 0; j < 8; ++j) {
        float v = acc[tm][tn][j] + bv;
        if (EPI == 1) v *= sv;
        const size_t idx = (size_t)(r0 + j) * ldc + c;
        if (EPI == 2) Cb[idx] = (__bf16)v;
        else          Cf[idx] = v;
      }
    }
  }
}

// ---------------------------------------------------------------------------
// Diagonal complex LRU scan: h_t = lam*h_{t-1} + u_t, per (b,h) channel.
// Emits hr and NEGATED hi as bf16 (bf16 WMMA has no A-negate; y uses (-hi)@Ci).
// ---------------------------------------------------------------------------
__global__ __launch_bounds__(256) void lru_scan_kernel(
    const float* __restrict__ ur, const float* __restrict__ ui,
    const float* __restrict__ nu_log, const float* __restrict__ theta_log,
    __bf16* __restrict__ hrb, __bf16* __restrict__ hib_neg) {
  const int tid = blockIdx.x * blockDim.x + threadIdx.x;   // 0 .. B*H-1
  const int b = tid >> 9;            // / H (512)
  const int h = tid & (Hv - 1);
  const float th = __expf(theta_log[h]);
  const float mag = __expf(-__expf(nu_log[h]));
  float sn, cs;
  __sincosf(th, &sn, &cs);
  const float lr = mag * cs;
  const float li = mag * sn;

  size_t idx = (size_t)b * Tv * Hv + h;
  float hr = 0.0f, hi = 0.0f;
  for (int t = 0; t < Tv; ++t, idx += Hv) {
    const float vr = ur[idx];
    const float vi = ui[idx];
    const float nhr = fmaf(lr, hr, fmaf(-li, hi, vr));
    const float nhi = fmaf(lr, hi, fmaf(li, hr, vi));
    hr = nhr; hi = nhi;
    hrb[idx]     = (__bf16)hr;
    hib_neg[idx] = (__bf16)(-hi);
  }
}

// ---------------------------------------------------------------------------
// Fused readout + LayerNorm:
//   y = hr@Cr + (-hi)@Ci + (cr-ci);  y = LN(y)*ln_scale + ln_bias -> bf16
// One block = 16 rows x full 512 cols. 8 waves x 4 N-tiles.
// Register double-buffered: k+32 fragments are fetched while the 8-WMMA
// chain for k executes, so loads overlap matrix ops instead of serializing.
// ---------------------------------------------------------------------------
#define YLD 516
__global__ __launch_bounds__(256) void cgemm_ln_kernel(
    const __bf16* __restrict__ hrb, const __bf16* __restrict__ hib_neg,
    const __bf16* __restrict__ Crt, const __bf16* __restrict__ Cit,
    const float* __restrict__ cr, const float* __restrict__ ci,
    const float* __restrict__ ln_scale, const float* __restrict__ ln_bias,
    __bf16* __restrict__ yb) {
  __shared__ float ytile[16][YLD];
  __shared__ float red[16][16][2];
  __shared__ float mstat[16][2];

  const int m0 = blockIdx.x * 16;
  const int w  = threadIdx.x >> 5;
  const int lane = threadIdx.x & 31;

  v8f acc[4] = {};

  // prologue: fragments for k = 0
  v16bf ar = load_frag_a(hrb,     Hv, m0, 0);
  v16bf ai = load_frag_a(hib_neg, Hv, m0, 0);
  v16bf bcr[4], bci[4];
#pragma unroll
  for (int t = 0; t < 4; ++t) {
    bcr[t] = load_frag_b(Crt, Hv, w * 64 + t * 16, 0);
    bci[t] = load_frag_b(Cit, Hv, w * 64 + t * 16, 0);
  }

  for (int k = 0; k < Hv; k += 32) {
    // issue next iteration's fragment loads first (overlap with WMMAs below)
    v16bf arn, ain, bcrn[4], bcin[4];
    const int kn = k + 32;
    if (kn < Hv) {
      arn = load_frag_a(hrb,     Hv, m0, kn);
      ain = load_frag_a(hib_neg, Hv, m0, kn);
#pragma unroll
      for (int t = 0; t < 4; ++t) {
        bcrn[t] = load_frag_b(Crt, Hv, w * 64 + t * 16, kn);
        bcin[t] = load_frag_b(Cit, Hv, w * 64 + t * 16, kn);
      }
    }

#pragma unroll
    for (int t = 0; t < 4; ++t) {
      acc[t] = wmma_bf16(ar, bcr[t], acc[t]);
      acc[t] = wmma_bf16(ai, bci[t], acc[t]);
    }

    if (kn < Hv) {
      ar = arn; ai = ain;
#pragma unroll
      for (int t = 0; t < 4; ++t) { bcr[t] = bcrn[t]; bci[t] = bcin[t]; }
    }
  }

  // spill accumulators (+ per-column constant) to LDS stripe
  const int cn = lane & 15;
  const int rh = lane >> 4;
#pragma unroll
  for (int t = 0; t < 4; ++t) {
    const int c = w * 64 + t * 16 + cn;
    const float cc = cr[c] - ci[c];
#pragma unroll
    for (int j = 0; j < 8; ++j)
      ytile[j + 8 * rh][c] = acc[t][j] + cc;
  }
  __syncthreads();

  // per-row partial sums (strided -> conflict-free across a wave)
  {
    const int row = threadIdx.x >> 4;
    const int seg = threadIdx.x & 15;
    float s = 0.0f, ss = 0.0f;
#pragma unroll 4
    for (int i = 0; i < 32; ++i) {
      const float v = ytile[row][seg + 16 * i];
      s += v; ss += v * v;
    }
    red[row][seg][0] = s;
    red[row][seg][1] = ss;
  }
  __syncthreads();
  if (threadIdx.x < 16) {
    float S = 0.0f, SS = 0.0f;
#pragma unroll
    for (int i = 0; i < 16; ++i) { S += red[threadIdx.x][i][0]; SS += red[threadIdx.x][i][1]; }
    const float mean = S * (1.0f / (float)Hv);
    const float var  = SS * (1.0f / (float)Hv) - mean * mean;
    mstat[threadIdx.x][0] = mean;
    mstat[threadIdx.x][1] = rsqrtf(var + LN_EPS);
  }
  __syncthreads();

  // normalize + affine + bf16 store (coalesced)
#pragma unroll 4
  for (int i = 0; i < 32; ++i) {
    const int idx = i * 256 + threadIdx.x;
    const int r = idx >> 9;
    const int c = idx & (Hv - 1);
    const float v = (ytile[r][c] - mstat[r][0]) * mstat[r][1] * ln_scale[c] + ln_bias[c];
    yb[(size_t)(m0 + r) * Hv + c] = (__bf16)v;
  }
}

// ---------------------------------------------------------------------------
// Host side
// ---------------------------------------------------------------------------
extern "C" void kernel_launch(void* const* d_in, const int* in_sizes, int n_in,
                              void* d_out, int out_size, void* d_ws, size_t ws_size,
                              hipStream_t stream) {
  const float* x         = (const float*)d_in[0];
  const float* nu_log    = (const float*)d_in[1];
  const float* theta_log = (const float*)d_in[2];
  const float* gamma_log = (const float*)d_in[3];
  const float* Br        = (const float*)d_in[4];
  const float* br        = (const float*)d_in[5];
  const float* Bi        = (const float*)d_in[6];
  const float* bi        = (const float*)d_in[7];
  const float* Cr        = (const float*)d_in[8];
  const float* cr        = (const float*)d_in[9];
  const float* Ci        = (const float*)d_in[10];
  const float* ci        = (const float*)d_in[11];
  const float* ln_scale  = (const float*)d_in[12];
  const float* ln_bias   = (const float*)d_in[13];
  const float* W1        = (const float*)d_in[14];
  const float* b1        = (const float*)d_in[15];
  const float* W2        = (const float*)d_in[16];
  const float* b2        = (const float*)d_in[17];
  float* out             = (float*)d_out;

  // workspace carve-out (256B aligned)
  char* ws = (char*)d_ws;
  size_t off = 0;
  auto alloc = [&](size_t bytes) -> void* {
    off = (off + 255) & ~(size_t)255;
    void* p = ws + off;
    off += bytes;
    return p;
  };
  __bf16* xb    = (__bf16*)alloc((size_t)Mv * Hv * 2);     // later reused as yb
  __bf16* Brt   = (__bf16*)alloc((size_t)Hv * Hv * 2);
  __bf16* Bit   = (__bf16*)alloc((size_t)Hv * Hv * 2);
  __bf16* Crt   = (__bf16*)alloc((size_t)Hv * Hv * 2);
  __bf16* Cit   = (__bf16*)alloc((size_t)Hv * Hv * 2);
  __bf16* W1t   = (__bf16*)alloc((size_t)H4v * Hv * 2);    // [N=2048, K=512]
  __bf16* W2t   = (__bf16*)alloc((size_t)Ov * H4v * 2);    // [N=512, K=2048]
  float*  gamma = (float*)alloc((size_t)Hv * 4);
  float*  ur    = (float*)alloc((size_t)Mv * Hv * 4);      // ur+ui contiguous,
  float*  ui    = (float*)alloc((size_t)Mv * Hv * 4);      // reused as zb (134MB)
  __bf16* hrb   = (__bf16*)alloc((size_t)Mv * Hv * 2);
  __bf16* hib   = (__bf16*)alloc((size_t)Mv * Hv * 2);
  __bf16* yb    = xb;                                      // alias (x done)
  __bf16* zb    = (__bf16*)ur;                             // alias (u done)
  (void)ws_size; (void)in_sizes; (void)n_in; (void)out_size;

  // 1. gamma = exp(gamma_log)
  prep_gamma_kernel<<<2, 256, 0, stream>>>(gamma_log, gamma, Hv);

  // 2. x -> bf16
  cvt_f32_bf16_kernel<<<2048, 256, 0, stream>>>(x, xb, (size_t)Mv * Hv / 4);

  // 3. transpose+convert weights: W[K,N] f32 -> Wt[N,K] bf16
  transpose_bf16_kernel<<<dim3(Hv / 32, Hv / 32), 256, 0, stream>>>(Br, Brt, Hv, Hv);
  transpose_bf16_kernel<<<dim3(Hv / 32, Hv / 32), 256, 0, stream>>>(Bi, Bit, Hv, Hv);
  transpose_bf16_kernel<<<dim3(Hv / 32, Hv / 32), 256, 0, stream>>>(Cr, Crt, Hv, Hv);
  transpose_bf16_kernel<<<dim3(Hv / 32, Hv / 32), 256, 0, stream>>>(Ci, Cit, Hv, Hv);
  transpose_bf16_kernel<<<dim3(Hv / 32, H4v / 32), 256, 0, stream>>>(W1, W1t, Hv, H4v);
  transpose_bf16_kernel<<<dim3(H4v / 32, Ov / 32), 256, 0, stream>>>(W2, W2t, H4v, Ov);

  // 4. u = gamma*(x@B + b): two GEMMs, M=32768, N=512, K=512
  {
    dim3 grid(Mv / BM, Hv / BN);
    wmma_gemm_kernel<1><<<grid, 256, 0, stream>>>(xb, Brt, br, gamma, ur, nullptr,
                                                  Hv, Hv, Hv, Hv);
    wmma_gemm_kernel<1><<<grid, 256, 0, stream>>>(xb, Bit, bi, gamma, ui, nullptr,
                                                  Hv, Hv, Hv, Hv);
  }

  // 5. sequential complex scan over T (8192 independent channels)
  lru_scan_kernel<<<(Bv * Hv) / 256, 256, 0, stream>>>(ur, ui, nu_log, theta_log,
                                                       hrb, hib);

  // 6. fused readout GEMMs + LayerNorm -> y bf16
  cgemm_ln_kernel<<<Mv / 16, 256, 0, stream>>>(hrb, hib, Crt, Cit, cr, ci,
                                               ln_scale, ln_bias, yb);

  // 7. z = y@W1 + b1 -> bf16, M=32768, N=2048, K=512
  {
    dim3 grid(Mv / BM, H4v / BN);
    wmma_gemm_kernel<2><<<grid, 256, 0, stream>>>(yb, W1t, b1, nullptr, nullptr, zb,
                                                  Hv, Hv, Hv, H4v);
  }

  // 8. out = z@W2 + b2 -> f32, M=32768, N=512, K=2048
  {
    dim3 grid(Mv / BM, Ov / BN);
    wmma_gemm_kernel<0><<<grid, 256, 0, stream>>>(zb, W2t, b2, nullptr, out, nullptr,
                                                  H4v, H4v, H4v, Ov);
  }
}
